// HGNNGConv_56315611185271
// MI455X (gfx1250) — matrix operand order repaired and checked by reading, not compiled, not run on hardware
//
#include <hip/hip_runtime.h>

// ---------------------------------------------------------------------------
// HGNN+ conv for MI455X (gfx1250, wave32).
//  K1: fused MLP  h = lrelu(x@W1+b1)@W2+b2  via v_wmma_f32_16x16x32_bf16,
//      all fragments fed by b128 LDS loads.
//  K2..K5: segment-mean scatters via global_atomic_add_f32 (L2-resident),
//      2 channels per thread (float2 gather + 2 atomics).
//  K6: fused divide + softmax-blend + leaky-relu epilogue (float4).
// ---------------------------------------------------------------------------

#define N_VERTS      100000
#define N_HYPEREDGES 10000
#define N_INC        1600000
#define N_EDGES      1600000
#define C_IN         256
#define C_MID        128
#define C_OUT        64
#define NEG_SLOPE    0.2f

typedef __bf16        v16bf  __attribute__((ext_vector_type(16)));
typedef float         v8f    __attribute__((ext_vector_type(8)));
typedef unsigned int  uint2v __attribute__((ext_vector_type(2)));
typedef unsigned int  uint4v __attribute__((ext_vector_type(4)));
typedef float         flt2v  __attribute__((ext_vector_type(2)));
typedef float         flt4v  __attribute__((ext_vector_type(4)));

union Frag { v16bf v; uint4v q[2]; };

__device__ __forceinline__ unsigned short f2bf(float f) {
    union { __bf16 b; unsigned short u; } cv;
    cv.b = (__bf16)f;
    return cv.u;
}
__device__ __forceinline__ unsigned int pack2bf(float lo, float hi) {
    return (unsigned int)f2bf(lo) | ((unsigned int)f2bf(hi) << 16);
}
__device__ __forceinline__ float lrelu(float f) { return f > 0.0f ? f : NEG_SLOPE * f; }

// ---------------- workspace layout (float units) ----------------
#define OFF_H      0u          // 100000*64
#define OFF_ESUM   6400000u    // 10000*64
#define OFF_EFEAT  7040000u    // 10000*64
#define OFF_HGSUM  7680000u    // 100000*64
#define OFF_GSUM   14080000u   // 100000*64
#define OFF_DEGE   20480000u   // 10000
#define OFF_DEGVH  20490000u   // 100000
#define OFF_DEGVG  20590000u   // 100000
#define OFF_W1T    20690000u   // 256*128 bf16 = 16384 floats, layout [n][k]
#define OFF_W2T    20706384u   // 128*64  bf16 =  4096 floats, layout [n][k]
#define ZERO_BASE  OFF_ESUM
#define ZERO_COUNT (OFF_W1T - OFF_ESUM)   // 14,290,000 floats (divisible by 4)

// ---------------- K0a: zero accumulators (b128 stores) ----------------
__global__ void zero_ws_kernel(float* __restrict__ ws) {
    unsigned int i = blockIdx.x * blockDim.x + threadIdx.x;   // quad index
    if (i < ZERO_COUNT / 4u) {
        flt4v z = {0.0f, 0.0f, 0.0f, 0.0f};
        *(flt4v*)(ws + ZERO_BASE + i * 4u) = z;
    }
}

// ---------------- K0b: W1/W2 -> bf16, transposed to [n][k] ----------------
__global__ void convert_w_kernel(const float* __restrict__ W1,
                                 const float* __restrict__ W2,
                                 unsigned short* __restrict__ w1t,
                                 unsigned short* __restrict__ w2t) {
    int i = blockIdx.x * blockDim.x + threadIdx.x;
    if (i < C_IN * C_MID) {                        // W1: [k=256][n=128] -> w1t[n][k]
        int k = i / C_MID, n = i % C_MID;
        w1t[n * C_IN + k] = f2bf(W1[i]);
    } else if (i < C_IN * C_MID + C_MID * C_OUT) { // W2: [k=128][n=64] -> w2t[n][k]
        int j = i - C_IN * C_MID;
        int k = j / C_OUT, n = j % C_OUT;
        w2t[n * C_MID + k] = f2bf(W2[j]);
    }
}

// ---------------- K1: fused two-layer MLP with WMMA ----------------
// block = 256 threads = 8 wave32, BM = 32 vertex rows per block.
#define BM 32
__global__ __launch_bounds__(256) void mlp_wmma_kernel(
    const float* __restrict__ x,
    const float* __restrict__ b1,
    const float* __restrict__ b2,
    const unsigned short* __restrict__ w1t,   // bf16 [128][256]
    const unsigned short* __restrict__ w2t,   // bf16 [64][128]
    float* __restrict__ h) {

    // bf16 pairs stored as uint (48 KB total LDS), 16B-aligned for b128 access
    __shared__ __align__(16) unsigned int sX  [BM * C_IN  / 2];    // [row][k]
    __shared__ __align__(16) unsigned int sW1 [C_MID * 32 / 2];    // K-slice [n][k32]
    __shared__ __align__(16) unsigned int sW2 [C_OUT * C_MID / 2]; // [n][k]
    __shared__ __align__(16) unsigned int sMid[BM * C_MID / 2];    // [row][k]

    const int tid  = threadIdx.x;
    const int wid  = tid >> 5;
    const int lane = tid & 31;
    const int hi   = lane >> 4;     // K-half selector per WMMA layout
    const int l16  = lane & 15;
    const int rowBlock = blockIdx.x * BM;

    // stage x tile: b128 global loads -> packed bf16 pairs (b64 LDS stores)
    {
        const flt4v* xg = (const flt4v*)(x + (size_t)rowBlock * C_IN);
        uint2v* sX2 = (uint2v*)sX;
        #pragma unroll
        for (int i = 0; i < 8; ++i) {
            int p = tid + i * 256;                 // 2048 float4
            flt4v f = xg[p];
            uint2v d;
            d.x = pack2bf(f.x, f.y);
            d.y = pack2bf(f.z, f.w);
            sX2[p] = d;
        }
    }
    // stage W2 fully: 1024 uint4
    {
        const uint4v* src = (const uint4v*)w2t;
        uint4v* dst = (uint4v*)sW2;
        #pragma unroll
        for (int i = 0; i < 4; ++i) dst[tid + i * 256] = src[tid + i * 256];
    }

    // ---- layer 1: [32 x 256] @ [256 x 128] ----
    const int rowTile  = wid >> 2;          // 0..1 (16 rows each)
    const int colBase  = (wid & 3) * 32;    // two 16-col tiles per wave
    const int aRow     = rowTile * 16 + l16;
    v8f c0 = {};
    v8f c1 = {};

    for (int k0 = 0; k0 < C_IN; k0 += 32) {
        __syncthreads();   // previous slice fully consumed
        // load W1 slice (512 uint4): all 128 cols, k in [k0,k0+32)
        {
            const uint4v* src = (const uint4v*)w1t;
            uint4v* dst = (uint4v*)sW1;
            #pragma unroll
            for (int i = 0; i < 2; ++i) {
                int q = tid + i * 256;             // 0..511
                int col = q >> 2;
                dst[q] = src[col * (C_IN / 8) + (k0 >> 3) + (q & 3)];
            }
        }
        __syncthreads();

        // A 16x32 bf16: u[v] holds K = hi*16 + 2v  -> two b128
        Frag a, bA, bB;
        const int abase = (aRow * C_IN + k0) >> 1;
        a.q[0] = *(const uint4v*)&sX[abase + hi * 8];
        a.q[1] = *(const uint4v*)&sX[abase + hi * 8 + 4];
        // B 32x16 bf16: u[v] holds K = hi*16 + 2v  -> two b128
        const int bbA = (colBase + l16) * 16 + hi * 8;
        bA.q[0] = *(const uint4v*)&sW1[bbA];
        bA.q[1] = *(const uint4v*)&sW1[bbA + 4];
        const int bbB = (colBase + 16 + l16) * 16 + hi * 8;
        bB.q[0] = *(const uint4v*)&sW1[bbB];
        bB.q[1] = *(const uint4v*)&sW1[bbB + 4];

        c0 = __builtin_amdgcn_wmma_f32_16x16x32_bf16(false, a.v, false, bA.v,
                                                     (short)0, c0, false, false);
        c1 = __builtin_amdgcn_wmma_f32_16x16x32_bf16(false, a.v, false, bB.v,
                                                     (short)0, c1, false, false);
    }

    // bias + leaky-relu, deposit bf16 activations to sMid
    {
        unsigned short* sMid16 = (unsigned short*)sMid;
        int n0 = colBase + l16, n1 = n0 + 16;
        float bias0 = b1[n0], bias1 = b1[n1];
        #pragma unroll
        for (int v = 0; v < 8; ++v) {
            int m = rowTile * 16 + hi * 8 + v;    // C/D layout: M = v + hi*8
            sMid16[m * C_MID + n0] = f2bf(lrelu(c0[v] + bias0));
            sMid16[m * C_MID + n1] = f2bf(lrelu(c1[v] + bias1));
        }
    }
    __syncthreads();

    // ---- layer 2: [32 x 128] @ [128 x 64] ----
    {
        const int rowT2 = wid >> 2;            // 0..1
        const int colT2 = (wid & 3) * 16;      // 0..48
        const int aRow2 = rowT2 * 16 + l16;
        const int ncol  = colT2 + l16;
        v8f c = {};
        #pragma unroll
        for (int k0 = 0; k0 < C_MID; k0 += 32) {
            Frag a, b;
            const int abase = aRow2 * (C_MID / 2) + (k0 >> 1);
            a.q[0] = *(const uint4v*)&sMid[abase + hi * 8];
            a.q[1] = *(const uint4v*)&sMid[abase + hi * 8 + 4];
            const int bbase = ncol * (C_MID / 2) + (k0 >> 1) + hi * 8;
            b.q[0] = *(const uint4v*)&sW2[bbase];
            b.q[1] = *(const uint4v*)&sW2[bbase + 4];
            c = __builtin_amdgcn_wmma_f32_16x16x32_bf16(false, a.v, false, b.v,
                                                        (short)0, c, false, false);
        }
        float bias = b2[ncol];
        int gr = rowBlock + rowT2 * 16;
        #pragma unroll
        for (int v = 0; v < 8; ++v) {
            int m = gr + hi * 8 + v;
            h[(size_t)m * C_OUT + ncol] = c[v] + bias;
        }
    }
}

// ---------------- K2: v2e scatter (2 channels / thread) ----------------
__global__ void v2e_kernel(const int* __restrict__ hg_v, const int* __restrict__ hg_e,
                           const float* __restrict__ h,
                           float* __restrict__ e_sum, float* __restrict__ deg_e) {
    int idx = blockIdx.x * blockDim.x + threadIdx.x;     // < N_INC*32
    int pair = idx >> 5, c2 = (idx & 31) * 2;
    int v = hg_v[pair], e = hg_e[pair];
    flt2v hv = *(const flt2v*)&h[(size_t)v * C_OUT + c2];
    float* dst = &e_sum[(size_t)e * C_OUT + c2];
    unsafeAtomicAdd(dst,     hv.x);
    unsafeAtomicAdd(dst + 1, hv.y);
    if (c2 == 0) unsafeAtomicAdd(&deg_e[e], 1.0f);
}

// ---------------- K3: hyperedge mean (float4) ----------------
__global__ void e_div_kernel(const float* __restrict__ e_sum, const float* __restrict__ deg_e,
                             float* __restrict__ e_feat) {
    int i = blockIdx.x * blockDim.x + threadIdx.x;       // < N_HYPEREDGES*16
    flt4v s = *(const flt4v*)&e_sum[i * 4];
    float inv = 1.0f / fmaxf(deg_e[(i * 4) >> 6], 1.0f);
    flt4v o = s * inv;
    *(flt4v*)&e_feat[i * 4] = o;
}

// ---------------- K4: e2v scatter (2 channels / thread) ----------------
__global__ void e2v_kernel(const int* __restrict__ hg_v, const int* __restrict__ hg_e,
                           const float* __restrict__ e_feat,
                           float* __restrict__ hg_sum, float* __restrict__ deg_vh) {
    int idx = blockIdx.x * blockDim.x + threadIdx.x;     // < N_INC*32
    int pair = idx >> 5, c2 = (idx & 31) * 2;
    int v = hg_v[pair], e = hg_e[pair];
    flt2v ef = *(const flt2v*)&e_feat[(size_t)e * C_OUT + c2];
    float* dst = &hg_sum[(size_t)v * C_OUT + c2];
    unsafeAtomicAdd(dst,     ef.x);
    unsafeAtomicAdd(dst + 1, ef.y);
    if (c2 == 0) unsafeAtomicAdd(&deg_vh[v], 1.0f);
}

// ---------------- K5: graph scatter (2 channels / thread) ----------------
__global__ void graph_kernel(const int* __restrict__ g_src, const int* __restrict__ g_dst,
                             const float* __restrict__ h,
                             float* __restrict__ g_sum, float* __restrict__ deg_vg) {
    int idx = blockIdx.x * blockDim.x + threadIdx.x;     // < N_EDGES*32
    int eidx = idx >> 5, c2 = (idx & 31) * 2;
    int s = g_src[eidx], d = g_dst[eidx];
    flt2v hv = *(const flt2v*)&h[(size_t)s * C_OUT + c2];
    float* dst = &g_sum[(size_t)d * C_OUT + c2];
    unsafeAtomicAdd(dst,     hv.x);
    unsafeAtomicAdd(dst + 1, hv.y);
    if (c2 == 0) unsafeAtomicAdd(&deg_vg[d], 1.0f);
}

// ---------------- K6: fused epilogue (float4) ----------------
__global__ void final_kernel(const float* __restrict__ w, const float* __restrict__ h,
                             const float* __restrict__ hg_sum, const float* __restrict__ g_sum,
                             const float* __restrict__ deg_vh, const float* __restrict__ deg_vg,
                             float* __restrict__ out) {
    int i = blockIdx.x * blockDim.x + threadIdx.x;       // < N_VERTS*16
    int base = i * 4;
    int v = base >> 6;
    float w0 = w[0], w1 = w[1];
    float m  = fmaxf(w0, w1);
    float e0 = __expf(w0 - m), e1 = __expf(w1 - m);
    float inv = 1.0f / (e0 + e1);
    float sw0 = e0 * inv, sw1 = e1 * inv;
    float idg = 1.0f / fmaxf(deg_vg[v], 1.0f);
    float idh = 1.0f / fmaxf(deg_vh[v], 1.0f);
    flt4v g  = *(const flt4v*)&g_sum[base];
    flt4v hg = *(const flt4v*)&hg_sum[base];
    flt4v hv = *(const flt4v*)&h[base];
    flt4v o;
    o.x = lrelu(sw0 * 0.5f * (g.x * idg + hg.x * idh) + sw1 * hv.x);
    o.y = lrelu(sw0 * 0.5f * (g.y * idg + hg.y * idh) + sw1 * hv.y);
    o.z = lrelu(sw0 * 0.5f * (g.z * idg + hg.z * idh) + sw1 * hv.z);
    o.w = lrelu(sw0 * 0.5f * (g.w * idg + hg.w * idh) + sw1 * hv.w);
    *(flt4v*)&out[base] = o;
}

// ---------------------------------------------------------------------------
extern "C" void kernel_launch(void* const* d_in, const int* in_sizes, int n_in,
                              void* d_out, int out_size, void* d_ws, size_t ws_size,
                              hipStream_t stream) {
    const float* x    = (const float*)d_in[0];
    const float* W1   = (const float*)d_in[1];
    const float* b1   = (const float*)d_in[2];
    const float* W2   = (const float*)d_in[3];
    const float* b2   = (const float*)d_in[4];
    const float* w    = (const float*)d_in[5];
    const int*   hg_v = (const int*)d_in[6];
    const int*   hg_e = (const int*)d_in[7];
    const int*   gsrc = (const int*)d_in[8];
    const int*   gdst = (const int*)d_in[9];
    float* out = (float*)d_out;
    float* ws  = (float*)d_ws;

    float* h       = ws + OFF_H;
    float* e_sum   = ws + OFF_ESUM;
    float* e_feat  = ws + OFF_EFEAT;
    float* hg_sum  = ws + OFF_HGSUM;
    float* g_sum   = ws + OFF_GSUM;
    float* deg_e   = ws + OFF_DEGE;
    float* deg_vh  = ws + OFF_DEGVH;
    float* deg_vg  = ws + OFF_DEGVG;
    unsigned short* w1t = (unsigned short*)(ws + OFF_W1T);
    unsigned short* w2t = (unsigned short*)(ws + OFF_W2T);

    // zero accumulators (one float4 / thread)
    {
        unsigned int quads = ZERO_COUNT / 4u;
        zero_ws_kernel<<<(quads + 255u) / 256u, 256, 0, stream>>>(ws);
    }
    // weight convert/transpose to bf16
    {
        int n = C_IN * C_MID + C_MID * C_OUT;   // 40960
        convert_w_kernel<<<(n + 255) / 256, 256, 0, stream>>>(W1, W2, w1t, w2t);
    }
    // fused MLP (WMMA)
    mlp_wmma_kernel<<<N_VERTS / BM, 256, 0, stream>>>(x, b1, b2, w1t, w2t, h);
    // hypergraph v2e
    v2e_kernel<<<(N_INC * 32) / 256, 256, 0, stream>>>(hg_v, hg_e, h, e_sum, deg_e);
    // hyperedge mean
    e_div_kernel<<<(N_HYPEREDGES * 16) / 256, 256, 0, stream>>>(e_sum, deg_e, e_feat);
    // graph scatter (independent of e_feat)
    graph_kernel<<<(N_EDGES * 32) / 256, 256, 0, stream>>>(gsrc, gdst, h, g_sum, deg_vg);
    // hypergraph e2v
    e2v_kernel<<<(N_INC * 32) / 256, 256, 0, stream>>>(hg_v, hg_e, e_feat, hg_sum, deg_vh);
    // fused epilogue
    final_kernel<<<(N_VERTS * 16) / 256, 256, 0, stream>>>(w, h, hg_sum, g_sum,
                                                           deg_vh, deg_vg, out);
}
